// MySAGE_89043261981499
// MI455X (gfx1250) — matrix operand order repaired and checked by reading, not compiled, not run on hardware
//
#include <hip/hip_runtime.h>
#include <math.h>

// Problem constants (from reference)
#define N_NODES 50000
#define IN_C    128
#define HID_C   256
#define OUT_C   128
#define N_EDGES 800000

typedef __attribute__((ext_vector_type(16))) _Float16 v16h;
typedef __attribute__((ext_vector_type(8)))  _Float16 v8h;
typedef __attribute__((ext_vector_type(2)))  _Float16 v2h;
typedef __attribute__((ext_vector_type(8)))  float    v8f;

// ---------------------------------------------------------------- utilities
__global__ void zero_f32(float* __restrict__ p, long n) {
  long i = (long)blockIdx.x * blockDim.x + threadIdx.x;
  long stride = (long)gridDim.x * blockDim.x;
  for (; i < n; i += stride) p[i] = 0.0f;
}

__global__ void degree_k(const int* __restrict__ edges, float* __restrict__ deg, int E) {
  int i = blockIdx.x * blockDim.x + threadIdx.x;
  int stride = gridDim.x * blockDim.x;
  for (; i < E; i += stride) {
    int dst = edges[E + i];           // edge_index row 1
    atomicAdd(&deg[dst], 1.0f);
  }
}

__global__ void invdeg_k(float* __restrict__ deg, int n) {
  int i = blockIdx.x * blockDim.x + threadIdx.x;
  if (i < n) deg[i] = 1.0f / fmaxf(deg[i], 1.0f);
}

// msg[dst, :] += x[src, :]   (float4-vectorized atomic scatter-add)
__global__ void scatter_k(const float* __restrict__ x, const int* __restrict__ edges,
                          float* __restrict__ msg, int E, int K) {
  int groups = K >> 2;                       // float4 groups per row
  long total = (long)E * groups;
  long i = (long)blockIdx.x * blockDim.x + threadIdx.x;
  long stride = (long)gridDim.x * blockDim.x;
  for (; i < total; i += stride) {
    int e = (int)(i / groups);
    int g = (int)(i % groups);
    int src = edges[e];
    int dst = edges[E + e];
    const float4 v = *reinterpret_cast<const float4*>(x + (size_t)src * K + 4 * g);
    float* o = msg + (size_t)dst * K + 4 * g;
    atomicAdd(o + 0, v.x);
    atomicAdd(o + 1, v.y);
    atomicAdd(o + 2, v.z);
    atomicAdd(o + 3, v.w);
  }
}

// ------------------------------------------------- fused SAGE layer (WMMA)
// out[M,N] = act( (Amsg .* invdeg_row) @ Wl  +  Aself @ Wr  +  bias )
//
// Block = 8 waves (256 threads), arranged waves_m x waves_n.
// Per k-step the block cooperatively stages:
//   lsA: (waves_m*16) x 32 f16, natural [row][k] layout (A frag k-groups
//        of 8 are contiguous -> 2x ds_load_b128 per fragment)
//   lsB: 32 x N f16, SWIZZLED as lsB[col*32 + k] so a lane's 16 B elements
//        (k = e + 16*half at fixed col) are contiguous -> 2x ds_load_b128.
// Each wave owns a 16x64 output strip: 1 A fragment reused by 4 WMMAs.
__global__ __launch_bounds__(256) void sage_gemm_k(
    const float* __restrict__ Amsg, const float* __restrict__ invdeg,
    const float* __restrict__ Aself,
    const float* __restrict__ Wl, const float* __restrict__ Wr,
    const float* __restrict__ bias, float* __restrict__ out,
    int M, int K, int N, int doRelu, int waves_m, int waves_n)
{
  __shared__ _Float16 lsA[64 * 32];    // up to 4 M-waves * 16 rows * 32 k (4 KB)
  __shared__ _Float16 lsB[256 * 32];   // up to N=256 cols * 32 k, swizzled (16 KB)

  const int wave = threadIdx.x >> 5;
  const int lane = threadIdx.x & 31;
  const int m2   = lane & 15;                 // local row / col within 16x16 tile
  const int half = lane >> 4;                 // lane half-group
  const int wm   = wave / waves_n;            // wave's M tile within block
  const int wn   = wave % waves_n;            // wave's 64-col group
  const int rowsM = waves_m * 16;             // rows staged per block
  const int r0   = blockIdx.x * rowsM;        // block's first output row

  v8f zero = {};
  v8f acc[4];
  acc[0] = zero; acc[1] = zero; acc[2] = zero; acc[3] = zero;

  for (int srcSel = 0; srcSel < 2; ++srcSel) {
    const float* A = srcSel ? Aself : Amsg;
    const float* W = srcSel ? Wr    : Wl;
    for (int k0 = 0; k0 < K; k0 += 32) {
      // ---- stage A macro-tile: rowsM x 32, f16 pairs (coalesced float2 reads)
      {
        int totalApairs = rowsM * 16;         // rowsM*32/2
        for (int p = threadIdx.x; p < totalApairs; p += 256) {
          int row = p >> 4;
          int kp  = (p & 15) << 1;
          int rg  = r0 + row;
          float2 v = make_float2(0.0f, 0.0f);
          float  s = 1.0f;
          if (rg < M) {
            v = *reinterpret_cast<const float2*>(A + (size_t)rg * K + k0 + kp);
            if (!srcSel) s = invdeg[rg];      // mean-aggregation scale
          }
          v2h h2;
          h2[0] = (_Float16)(v.x * s);
          h2[1] = (_Float16)(v.y * s);
          *reinterpret_cast<v2h*>(&lsA[row * 32 + kp]) = h2;
        }
      }
      // ---- stage B panel: 32 x N, swizzled lsB[c*32 + k] (k-contiguous per col)
      {
        int totalBpairs = N * 16;             // N*32/2
        for (int p = threadIdx.x; p < totalBpairs; p += 256) {
          int c  = p % N;                     // consecutive threads -> coalesced
          int kp = (p / N) << 1;
          float b0 = W[(size_t)(k0 + kp) * N + c];
          float b1 = W[(size_t)(k0 + kp + 1) * N + c];
          v2h h2;
          h2[0] = (_Float16)b0;
          h2[1] = (_Float16)b1;
          *reinterpret_cast<v2h*>(&lsB[c * 32 + kp]) = h2;
        }
      }
      // speculative prefetch of next A k-tile (global_prefetch_b8)
      if (k0 + 32 < K) {
        int prow = r0 + (int)(threadIdx.x >> 3);
        if (prow < M)
          __builtin_prefetch(A + (size_t)prow * K + k0 + 32 + ((threadIdx.x & 7) << 2), 0, 1);
      }
      __syncthreads();

      // ---- A fragment (k = 16*half + e, matching B): two contiguous 8-element
      //      k-groups per lane -> 2x ds_load_b128
      const _Float16* aBase = &lsA[(wm * 16 + m2) * 32];
      v8h alo = *reinterpret_cast<const v8h*>(aBase + 16 * half);
      v8h ahi = *reinterpret_cast<const v8h*>(aBase + 16 * half + 8);
      v16h a;
      for (int e = 0; e < 8; ++e) { a[e] = alo[e]; a[e + 8] = ahi[e]; }

      // ---- 4 WMMAs reuse the A fragment across the wave's 64-column strip
      for (int j = 0; j < 4; ++j) {
        int col = wn * 64 + j * 16 + m2;
        v16h b = *reinterpret_cast<const v16h*>(&lsB[col * 32 + 16 * half]);
        acc[j] = __builtin_amdgcn_wmma_f32_16x16x32_f16(
            /*neg_a=*/false, a, /*neg_b=*/false, b,
            /*c_mod=*/(short)0, acc[j], /*reuse_a=*/false, /*reuse_b=*/false);
      }
      __syncthreads();
    }
  }

  // ---- epilogue: C layout VGPR v -> row (v + 8*half), col = lane&15
  for (int j = 0; j < 4; ++j) {
    int n  = wn * 64 + j * 16 + m2;
    float bv = bias[n];
    for (int v = 0; v < 8; ++v) {
      int r = r0 + wm * 16 + v + 8 * half;
      if (r < M) {
        float val = acc[j][v] + bv;
        if (doRelu) val = fmaxf(val, 0.0f);
        out[(size_t)r * N + n] = val;
      }
    }
  }
}

// ---------------------------------------------- link decode: sigmoid(<z_u,z_v>)
// wave32 per edge; 128 channels = 32 lanes x float4 (b128 loads)
__global__ void edge_dot_k(const float* __restrict__ z, const int* __restrict__ edges,
                           float* __restrict__ out, int E) {
  int waveInBlock = threadIdx.x >> 5;
  int lane        = threadIdx.x & 31;
  int e = blockIdx.x * (blockDim.x >> 5) + waveInBlock;
  if (e >= E) return;
  int src = edges[e];
  int dst = edges[E + e];
  const float4 a = reinterpret_cast<const float4*>(z + (size_t)src * OUT_C)[lane];
  const float4 b = reinterpret_cast<const float4*>(z + (size_t)dst * OUT_C)[lane];
  float d = a.x * b.x + a.y * b.y + a.z * b.z + a.w * b.w;
  for (int off = 16; off > 0; off >>= 1) d += __shfl_xor(d, off, 32);
  if (lane == 0) out[e] = 1.0f / (1.0f + expf(-d));
}

// ------------------------------------------------------------------ driver
extern "C" void kernel_launch(void* const* d_in, const int* in_sizes, int n_in,
                              void* d_out, int out_size, void* d_ws, size_t ws_size,
                              hipStream_t stream) {
  (void)in_sizes; (void)n_in; (void)out_size; (void)ws_size;

  const float* x    = (const float*)d_in[0];
  const int*   edge = (const int*)  d_in[1];   // [2, E] row-major
  const float* w1l  = (const float*)d_in[2];
  const float* b1   = (const float*)d_in[3];
  const float* w1r  = (const float*)d_in[4];
  const float* w2l  = (const float*)d_in[5];
  const float* b2   = (const float*)d_in[6];
  const float* w2r  = (const float*)d_in[7];
  float* out = (float*)d_out;

  // workspace layout (floats): deg | msg(50000x256, reused) | h(50000x256) | z(50000x128)
  float* ws  = (float*)d_ws;
  float* deg = ws;
  float* msg = ws + 50048;
  float* h   = msg + (size_t)N_NODES * HID_C;
  float* z   = h   + (size_t)N_NODES * HID_C;

  // degrees (shared by both layers)
  zero_f32<<<1024, 256, 0, stream>>>(deg, (long)N_NODES);
  degree_k<<<2048, 256, 0, stream>>>(edge, deg, N_EDGES);
  invdeg_k<<<(N_NODES + 255) / 256, 256, 0, stream>>>(deg, N_NODES);

  // ---- layer 1: h = relu(mean_agg(x) @ w1_l + x @ w1_r + b1)
  zero_f32<<<4096, 256, 0, stream>>>(msg, (long)N_NODES * IN_C);
  scatter_k<<<8192, 256, 0, stream>>>(x, edge, msg, N_EDGES, IN_C);
  {
    // waves 2(M) x 4(N): block tile 32 x 256
    int rowsPerBlock = 2 * 16;
    int blocks = (N_NODES + rowsPerBlock - 1) / rowsPerBlock;   // 1563
    sage_gemm_k<<<blocks, 256, 0, stream>>>(msg, deg, x, w1l, w1r, b1, h,
                                            N_NODES, IN_C, HID_C, /*relu=*/1,
                                            /*waves_m=*/2, /*waves_n=*/4);
  }

  // ---- layer 2: z = mean_agg(h) @ w2_l + h @ w2_r + b2
  zero_f32<<<8192, 256, 0, stream>>>(msg, (long)N_NODES * HID_C);
  scatter_k<<<8192, 256, 0, stream>>>(h, edge, msg, N_EDGES, HID_C);
  {
    // waves 4(M) x 2(N): block tile 64 x 128
    int rowsPerBlock = 4 * 16;
    int blocks = (N_NODES + rowsPerBlock - 1) / rowsPerBlock;   // 782
    sage_gemm_k<<<blocks, 256, 0, stream>>>(msg, deg, h, w2l, w2r, b2, z,
                                            N_NODES, HID_C, OUT_C, /*relu=*/0,
                                            /*waves_m=*/4, /*waves_n=*/2);
  }

  // ---- link decode
  edge_dot_k<<<(N_EDGES + 7) / 8, 256, 0, stream>>>(z, edge, out, N_EDGES);
}
